// MultiHeadAttention_9887014715580
// MI455X (gfx1250) — matrix operand order repaired and checked by reading, not compiled, not run on hardware
//
#include <hip/hip_runtime.h>

// ---------- types ----------
typedef __bf16 bf16_t;
typedef __attribute__((ext_vector_type(16))) __bf16 v16bf;
typedef __attribute__((ext_vector_type(8)))  __bf16 v8bf;
typedef __attribute__((ext_vector_type(4)))  __bf16 v4bf;
typedef __attribute__((ext_vector_type(8)))  float  v8f;
typedef __attribute__((ext_vector_type(4)))  float  v4f;

__device__ __forceinline__ v16bf join16(v8bf lo, v8bf hi) {
    return __builtin_shufflevector(lo, hi, 0, 1, 2, 3, 4, 5, 6, 7,
                                           8, 9, 10, 11, 12, 13, 14, 15);
}

// LDS offset of a __shared__ object (generic ptr low 32 bits = LDS offset)
__device__ __forceinline__ unsigned lds_off(const void* p) {
    return (unsigned)(reinterpret_cast<uintptr_t>(p));
}

// CDNA5 async global->LDS copy, 16 bytes per active lane (tracked by ASYNCcnt)
__device__ __forceinline__ void async_b128(unsigned lds_byte_off, const void* g) {
    asm volatile("global_load_async_to_lds_b128 %0, %1, off"
                 :: "v"(lds_byte_off), "v"((unsigned long long)(uintptr_t)g)
                 : "memory");
}
__device__ __forceinline__ void wait_asynccnt0() {
    asm volatile("s_wait_asynccnt 0x0" ::: "memory");
}

// one-instruction 32-lane XOR butterfly via ds_swizzle (group-of-32 pattern)
template <int MASK>
__device__ __forceinline__ float swz_xor(float x) {
    return __int_as_float(
        __builtin_amdgcn_ds_swizzle(__float_as_int(x), (MASK << 10) | 0x1f));
}

// batched 8-row max reduction over 16-lane groups: per XOR stage, all 8
// ds_swizzles issue together (clause-able, one dscnt wait) then 8 v_max.
__device__ __forceinline__ void redmax16_8(float (&x)[8]) {
    float tmp[8];
    #pragma unroll
    for (int i = 0; i < 8; ++i) tmp[i] = swz_xor<1>(x[i]);
    #pragma unroll
    for (int i = 0; i < 8; ++i) x[i] = fmaxf(x[i], tmp[i]);
    #pragma unroll
    for (int i = 0; i < 8; ++i) tmp[i] = swz_xor<2>(x[i]);
    #pragma unroll
    for (int i = 0; i < 8; ++i) x[i] = fmaxf(x[i], tmp[i]);
    #pragma unroll
    for (int i = 0; i < 8; ++i) tmp[i] = swz_xor<4>(x[i]);
    #pragma unroll
    for (int i = 0; i < 8; ++i) x[i] = fmaxf(x[i], tmp[i]);
    #pragma unroll
    for (int i = 0; i < 8; ++i) tmp[i] = swz_xor<8>(x[i]);
    #pragma unroll
    for (int i = 0; i < 8; ++i) x[i] = fmaxf(x[i], tmp[i]);
}

// ---------- fp32 -> bf16 convert (4-wide) ----------
__global__ void mha_f32_to_bf16(const float* __restrict__ in,
                                bf16_t* __restrict__ out, size_t n /* %4==0 */) {
    size_t i = (size_t)blockIdx.x * blockDim.x + threadIdx.x;
    size_t stride = (size_t)gridDim.x * blockDim.x;
    for (size_t e = i * 4; e < n; e += stride * 4) {
        v4f v = *(const v4f*)(in + e);
        *(v4bf*)(out + e) = __builtin_convertvector(v, v4bf);
    }
}

// ---------- tiled GEMM: C[M,N] = A[M,K]@B[K,N] + bias ----------
// 256 threads = 8 waves (4m x 2n); block tile 128(M) x 128(N), K-step 32.
// Each wave: 32x64 -> 2x4 WMMA accumulators = 8 WMMAs per K-step.
// Double-buffered LDS; A tile staged with async global->LDS DMA issued
// before the current tile's WMMAs so the copy hides under compute.
template <typename OUT_T>
__global__ __launch_bounds__(256)
void mha_gemm_bf16(const bf16_t* __restrict__ A, const bf16_t* __restrict__ Bm,
                   const float* __restrict__ bias, OUT_T* __restrict__ C,
                   int M, int N, int K) {
    __shared__ bf16_t As[2][128][32];    // [buf][m][k]      16 KB
    __shared__ bf16_t BsT[2][128][32];   // [buf][n][k] (T)  16 KB

    const int t    = threadIdx.x;
    const int wave = t >> 5;
    const int lane = t & 31;
    const int wm   = wave & 3;
    const int wn   = wave >> 2;
    const int l16  = lane & 15;
    const int hl   = lane >> 4;

    const int row0 = blockIdx.y * 128;
    const int col0 = blockIdx.x * 128;

    const int sm_a = t >> 1;             // A-stage row 0..127
    const int bo_a = (t & 1) * 32;       // A-stage byte offset in 64B row
    const int sk_b = t >> 3;             // B-stage k 0..31
    const int sn_b = (t & 7) * 16;       // B-stage n offset 0..112

    const unsigned as_base = lds_off(&As[0][0][0]);

    v8f acc[2][4] = {};

    auto stage = [&](int buf, int kk0) {
        // A: straight row-major copy -> async DMA (2 x b128 per thread)
        {
            const bf16_t* asrc = A + (size_t)(row0 + sm_a) * K + kk0 + bo_a / 2;
            unsigned ldst = as_base + (unsigned)buf * 8192u + sm_a * 64 + bo_a;
            async_b128(ldst, asrc);
            async_b128(ldst + 16, asrc + 8);
        }
        // B: transpose on the fly (contiguous global read, strided LDS write)
        {
            const bf16_t* bsrc = Bm + (size_t)(kk0 + sk_b) * N + col0 + sn_b;
            v8bf b0 = *(const v8bf*)(bsrc);
            v8bf b1 = *(const v8bf*)(bsrc + 8);
            #pragma unroll
            for (int j = 0; j < 8; ++j) {
                BsT[buf][sn_b + j][sk_b]     = b0[j];
                BsT[buf][sn_b + 8 + j][sk_b] = b1[j];
            }
        }
    };

    stage(0, 0);
    wait_asynccnt0();
    __syncthreads();

    int cur = 0;
    for (int k0 = 0; k0 < K; k0 += 32, cur ^= 1) {
        if (k0 + 32 < K) stage(cur ^ 1, k0 + 32);   // overlaps with WMMAs

        const int kba = hl ? 16 : 0;
        v16bf afrag[2];
        #pragma unroll
        for (int sm = 0; sm < 2; ++sm) {
            const int m = wm * 32 + sm * 16 + l16;
            v8bf lo = *(const v8bf*)&As[cur][m][kba];
            v8bf hi = *(const v8bf*)&As[cur][m][kba + 8];
            afrag[sm] = join16(lo, hi);
        }
        const int kbb = hl ? 16 : 0;
        v16bf bfrag[4];
        #pragma unroll
        for (int sn = 0; sn < 4; ++sn) {
            const int n = wn * 64 + sn * 16 + l16;
            v8bf lo = *(const v8bf*)&BsT[cur][n][kbb];
            v8bf hi = *(const v8bf*)&BsT[cur][n][kbb + 8];
            bfrag[sn] = join16(lo, hi);
        }
        #pragma unroll
        for (int sm = 0; sm < 2; ++sm)
            #pragma unroll
            for (int sn = 0; sn < 4; ++sn)
                acc[sm][sn] = __builtin_amdgcn_wmma_f32_16x16x32_bf16(
                    false, afrag[sm], false, bfrag[sn],
                    (short)0, acc[sm][sn], false, false);

        wait_asynccnt0();
        __syncthreads();
    }

    #pragma unroll
    for (int sm = 0; sm < 2; ++sm) {
        #pragma unroll
        for (int sn = 0; sn < 4; ++sn) {
            const int n  = col0 + wn * 64 + sn * 16 + l16;
            const float bv = bias[n];
            const int mb = row0 + wm * 32 + sm * 16 + hl * 8;
            #pragma unroll
            for (int i = 0; i < 8; ++i)
                C[(size_t)(mb + i) * N + n] = (OUT_T)(acc[sm][sn][i] + bv);
        }
    }
}

// ---------- flash attention, causal, bf16 WMMA ----------
// grid: (B*H, S/128); block: 256 threads = 8 waves; wave owns 16 query rows.
// Row-sum of P via WMMA against an all-ones B fragment. Causal masking and
// skipping the second key-subtile's WMMAs happen only on diagonal tiles (wave-uniform).
__global__ __launch_bounds__(256)
void mha_flash_attn(const bf16_t* __restrict__ qkv,  // [B*S, 3*D]; per head q|k|v of 64
                    bf16_t* __restrict__ out,        // [B*S, D]
                    int Bn, int S, int H) {
    const int DEPTH = 64;
    const int D3 = 3 * H * DEPTH;      // 3072
    const int Dd = H * DEPTH;          // 1024

    const int bh = blockIdx.x;
    const int bi = bh / H, h = bh % H;
    const int blockq0 = blockIdx.y * 128;

    const int t    = threadIdx.x;
    const int wave = t >> 5;
    const int lane = t & 31;
    const int l16  = lane & 15;
    const int hl   = lane >> 4;
    const int qrow0 = blockq0 + wave * 16;

    __shared__ bf16_t Ks[32][64];      // [kv][depth] row-major (async DMA) 4 KB
    __shared__ bf16_t VsT[64][32];     // [depth][kv] transposed            4 KB
    __shared__ bf16_t Ps[8][16][32];   // per-wave P staging                8 KB

    const float NEG_INF = -__builtin_inff();
    const float cscale  = 0.125f * 1.44269504088896f;  // 1/sqrt(64) * log2(e)

    const unsigned ks_base = lds_off(&Ks[0][0]);

    // all-ones B fragment for row-sum WMMA
    v16bf ones;
    #pragma unroll
    for (int j = 0; j < 16; ++j) ones[j] = (bf16_t)1.0f;

    // ---- Q A-fragments (held in registers for the whole kv loop) ----
    const int qtok = qrow0 + l16;
    const bf16_t* qbase = qkv + (size_t)(bi * S + qtok) * D3 + h * 3 * DEPTH;
    const int kba = hl ? 16 : 0;
    v16bf qa[2];
    #pragma unroll
    for (int kk = 0; kk < 2; ++kk) {
        v8bf lo = *(const v8bf*)(qbase + kk * 32 + kba);
        v8bf hi = *(const v8bf*)(qbase + kk * 32 + kba + 8);
        qa[kk] = join16(lo, hi);
    }

    v8f O[4] = {};
    float mrow[8], lrow[8];
    #pragma unroll
    for (int i = 0; i < 8; ++i) { mrow[i] = NEG_INF; lrow[i] = 0.f; }

    const int sk_r  = t >> 3;          // K/V-stage token 0..31
    const int sk_bo = (t & 7) * 16;    // K-stage byte offset in 128B row
    const int sv_d  = (t & 7) * 8;     // V-stage depth offset

    const int kvend = blockq0 + 128;   // block-uniform trip count
    for (int kv0 = 0; kv0 < kvend; kv0 += 32) {
        // ---- K tile: straight copy via async DMA (B-frag layout = row-major) ----
        {
            const bf16_t* ksrc = qkv + (size_t)(bi * S + kv0 + sk_r) * D3
                               + h * 3 * DEPTH + DEPTH + sk_bo / 2;
            async_b128(ks_base + sk_r * 128 + sk_bo, ksrc);
        }
        // ---- V tile: transpose on the fly (overlaps with async K) ----
        {
            const bf16_t* vsrc = qkv + (size_t)(bi * S + kv0 + sk_r) * D3
                               + h * 3 * DEPTH + 2 * DEPTH + sv_d;
            v8bf v0 = *(const v8bf*)(vsrc);
            #pragma unroll
            for (int j = 0; j < 8; ++j) VsT[sv_d + j][sk_r] = v0[j];
        }
        wait_asynccnt0();
        __syncthreads();

        if (kv0 < qrow0 + 16) {        // causal: wave needs keys <= qrow0+15
            const bool have2    = (kv0 + 16 <= qrow0 + 15); // 2nd subtile has any valid key
            const bool diagonal = (kv0 + 31 > qrow0);       // any masked entry in tile

            // ---- scores; second subtile's WMMAs skipped above the diagonal ----
            const int kbb = hl ? 16 : 0;
            v8f sc[2] = {};
            #pragma unroll
            for (int kk = 0; kk < 2; ++kk) {
                const bf16_t* kp = &Ks[l16][kk * 32 + kbb];
                v16bf kb16 = join16(*(const v8bf*)kp, *(const v8bf*)(kp + 8));
                sc[0] = __builtin_amdgcn_wmma_f32_16x16x32_bf16(
                    false, qa[kk], false, kb16, (short)0, sc[0], false, false);
            }
            if (have2) {
                #pragma unroll
                for (int kk = 0; kk < 2; ++kk) {
                    const bf16_t* kp = &Ks[16 + l16][kk * 32 + kbb];
                    v16bf kb16 = join16(*(const v8bf*)kp, *(const v8bf*)(kp + 8));
                    sc[1] = __builtin_amdgcn_wmma_f32_16x16x32_bf16(
                        false, qa[kk], false, kb16, (short)0, sc[1], false, false);
                }
            }

            // ---- mask (diagonal tiles only) + batched max butterfly ----
            float s0[8], s1[8];
            if (!diagonal) {
                #pragma unroll
                for (int i = 0; i < 8; ++i) { s0[i] = sc[0][i]; s1[i] = sc[1][i]; }
            } else {
                #pragma unroll
                for (int i = 0; i < 8; ++i) {
                    const int qr = qrow0 + hl * 8 + i;
                    s0[i] = (kv0 + l16      <= qr) ? sc[0][i] : NEG_INF;
                    s1[i] = (have2 && kv0 + 16 + l16 <= qr) ? sc[1][i] : NEG_INF;
                }
            }
            float mx[8];
            #pragma unroll
            for (int i = 0; i < 8; ++i) mx[i] = fmaxf(s0[i], s1[i]);
            redmax16_8(mx);

            float p0[8], p1[8];
            #pragma unroll
            for (int i = 0; i < 8; ++i) {
                const float mnew  = fmaxf(mrow[i], mx[i]);
                const float alpha = __builtin_amdgcn_exp2f((mrow[i] - mnew) * cscale);
                p0[i] = __builtin_amdgcn_exp2f((s0[i] - mnew) * cscale);
                p1[i] = __builtin_amdgcn_exp2f((s1[i] - mnew) * cscale);
                mrow[i] = mnew;
                lrow[i] *= alpha;
                #pragma unroll
                for (int tt = 0; tt < 4; ++tt) O[tt][i] *= alpha;
            }

            // ---- P: C-layout -> A-layout via per-wave LDS bounce ----
            #pragma unroll
            for (int i = 0; i < 8; ++i) {
                Ps[wave][hl * 8 + i][l16]      = (bf16_t)p0[i];
                Ps[wave][hl * 8 + i][16 + l16] = (bf16_t)p1[i];
            }
            v16bf pa;
            {
                v8bf lo = *(const v8bf*)&Ps[wave][l16][kba];
                v8bf hi = *(const v8bf*)&Ps[wave][l16][kba + 8];
                pa = join16(lo, hi);
            }

            // ---- row sums via WMMA against all-ones B (broadcast in C-layout) ----
            {
                v8f z = {};
                v8f rs = __builtin_amdgcn_wmma_f32_16x16x32_bf16(
                    false, pa, false, ones, (short)0, z, false, false);
                #pragma unroll
                for (int i = 0; i < 8; ++i) lrow[i] += rs[i];
            }

            // ---- PV: 4 WMMAs over depth N-tiles ----
            const int kbv = hl ? 16 : 0;
            #pragma unroll
            for (int tt = 0; tt < 4; ++tt) {
                const int n = tt * 16 + l16;
                v8bf lo = *(const v8bf*)&VsT[n][kbv];
                v8bf hi = *(const v8bf*)&VsT[n][kbv + 8];
                v16bf vb = join16(lo, hi);
                O[tt] = __builtin_amdgcn_wmma_f32_16x16x32_bf16(
                    false, pa, false, vb, (short)0, O[tt], false, false);
            }
        }
        __syncthreads();
    }

    // ---- normalize and store bf16 [b, s, h*64 + d] ----
    #pragma unroll
    for (int i = 0; i < 8; ++i) {
        const int srow = qrow0 + hl * 8 + i;
        const float inv = 1.0f / lrow[i];
        bf16_t* orow = out + (size_t)(bi * S + srow) * Dd + h * DEPTH;
        #pragma unroll
        for (int tt = 0; tt < 4; ++tt)
            orow[tt * 16 + l16] = (bf16_t)(O[tt][i] * inv);
    }
}

// ---------- host launcher ----------
extern "C" void kernel_launch(void* const* d_in, const int* in_sizes, int n_in,
                              void* d_out, int out_size, void* d_ws, size_t ws_size,
                              hipStream_t stream) {
    (void)in_sizes; (void)n_in; (void)out_size; (void)ws_size;
    const int Bn = 2, S = 2048, D = 1024, H = 16;
    const size_t M = (size_t)Bn * S;        // 4096

    const float* x     = (const float*)d_in[0];
    // d_in[1] = causal mask (implicit in kernel)
    const float* Wqkv  = (const float*)d_in[2];
    const float* bqkv  = (const float*)d_in[3];
    const float* Wout  = (const float*)d_in[4];
    const float* bout  = (const float*)d_in[5];
    float* outp        = (float*)d_out;

    char* ws = (char*)d_ws;
    size_t off = 0;
    bf16_t* xb    = (bf16_t*)(ws + off); off += M * D * sizeof(bf16_t);
    bf16_t* wqkvb = (bf16_t*)(ws + off); off += (size_t)D * 3 * D * sizeof(bf16_t);
    bf16_t* woutb = (bf16_t*)(ws + off); off += (size_t)D * D * sizeof(bf16_t);
    bf16_t* qkvb  = (bf16_t*)(ws + off); off += M * 3 * D * sizeof(bf16_t);
    bf16_t* attb  = (bf16_t*)(ws + off); off += M * D * sizeof(bf16_t);

    // 1) converts (4-wide)
    mha_f32_to_bf16<<<2048, 256, 0, stream>>>(x,    xb,    M * D);
    mha_f32_to_bf16<<<2048, 256, 0, stream>>>(Wqkv, wqkvb, (size_t)D * 3 * D);
    mha_f32_to_bf16<<<1024, 256, 0, stream>>>(Wout, woutb, (size_t)D * D);

    // 2) QKV projection: [4096,1024] @ [1024,3072] + b -> bf16
    {
        dim3 g(3 * D / 128, (unsigned)(M / 128));
        mha_gemm_bf16<bf16_t><<<g, 256, 0, stream>>>(xb, wqkvb, bqkv, qkvb,
                                                     (int)M, 3 * D, D);
    }

    // 3) flash attention per (b,h), 128 query rows per block
    {
        dim3 g(Bn * H, S / 128);
        mha_flash_attn<<<g, 256, 0, stream>>>(qkvb, attb, Bn, S, H);
    }

    // 4) output projection: [4096,1024] @ [1024,1024] + b -> fp32 d_out
    {
        dim3 g(D / 128, (unsigned)(M / 128));
        mha_gemm_bf16<float><<<g, 256, 0, stream>>>(attb, woutb, bout, outp,
                                                    (int)M, D, D);
    }
}
